// EdgeEnhancedAPPNP_54382875902207
// MI455X (gfx1250) — compile-verified, matrix operand order checked by reading, not run on hardware
//
#include <hip/hip_runtime.h>

typedef __attribute__((ext_vector_type(16))) _Float16 v16h;
typedef __attribute__((ext_vector_type(8)))  _Float16 v8h;
typedef __attribute__((ext_vector_type(8)))  float    v8f;

#define N_NODES 100000
#define N_EDGES 1250000
#define IN_C    64
#define HID     128
#define OUT_C   64
#define K_ITER  10
#define ALPHA_F 0.1f

// ---------------- fragment builders (wave32, v_wmma_f32_16x16x32_f16) ----------------
// A (16x32 tile of f32 row-major A, stride lda): lane = {m = lane&15, khalf = (lane>>4)*8}
//   a[t] = A[m][k0+khalf+t], a[t+8] = A[m][k0+16+khalf+t].  Vectorized: 4x b128 loads + cvt.
__device__ __forceinline__ v16h load_a_frag_f32(const float* __restrict__ A, int lda,
                                                int k0, int lane) {
  int m  = lane & 15;
  int kb = (lane >> 4) << 3;
  const float* arow = A + (size_t)m * lda + k0 + kb;
  float4 x0 = *(const float4*)(arow);
  float4 x1 = *(const float4*)(arow + 4);
  float4 y0 = *(const float4*)(arow + 16);
  float4 y1 = *(const float4*)(arow + 20);
  v16h a;
  a[0]  = (_Float16)x0.x; a[1]  = (_Float16)x0.y; a[2]  = (_Float16)x0.z; a[3]  = (_Float16)x0.w;
  a[4]  = (_Float16)x1.x; a[5]  = (_Float16)x1.y; a[6]  = (_Float16)x1.z; a[7]  = (_Float16)x1.w;
  a[8]  = (_Float16)y0.x; a[9]  = (_Float16)y0.y; a[10] = (_Float16)y0.z; a[11] = (_Float16)y0.w;
  a[12] = (_Float16)y1.x; a[13] = (_Float16)y1.y; a[14] = (_Float16)y1.z; a[15] = (_Float16)y1.w;
  return a;
}

// B (32x16 tile of W[K x NC]) read from TRANSPOSED f16 weights Wt[NC x K]:
//   b[t] = W[k0+kb+t][n] = Wt[n][k0+kb+t]  -> two contiguous 16B (b128) loads per lane.
__device__ __forceinline__ v16h load_b_fragT(const _Float16* __restrict__ Wt, int ldk,
                                             int k0, int n0, int lane) {
  int n  = n0 + (lane & 15);
  int kb = (lane >> 4) << 3;
  const _Float16* wc = Wt + (size_t)n * ldk + k0 + kb;
  v8h lo = *(const v8h*)(wc);
  v8h hi = *(const v8h*)(wc + 16);
  v16h b;
#pragma unroll
  for (int t = 0; t < 8; ++t) { b[t] = lo[t]; b[t + 8] = hi[t]; }
  return b;
}

// ---------------- generic GEMM: Out[M,NC] = A[M,K] @ W[K,NC] + bias (opt. relu) --------------
// One wave per 16-row strip: A fragments (K/32 of them) live in registers and are reused
// across every NC/16 column tile -> A is read exactly once from HBM.
template <int K, int NC, int RELU>
__global__ void gemm_wmma_f16(const float* __restrict__ A, const _Float16* __restrict__ Wt,
                              const float* __restrict__ bias, float* __restrict__ Out, int M) {
  int wave = blockIdx.x * (blockDim.x >> 5) + (threadIdx.x >> 5);
  int lane = threadIdx.x & 31;
  int m0 = wave << 4;
  if (m0 >= M) return;

  constexpr int NK = K / 32;
  v16h afr[NK];
#pragma unroll
  for (int kk = 0; kk < NK; ++kk)
    afr[kk] = load_a_frag_f32(A + (size_t)m0 * K, K, kk << 5, lane);

  int n  = lane & 15;
  int mr = (lane >> 4) << 3;
#pragma unroll
  for (int nt = 0; nt < NC / 16; ++nt) {
    int n0 = nt << 4;
    v8f acc = {};
#pragma unroll
    for (int kk = 0; kk < NK; ++kk) {
      v16h b = load_b_fragT(Wt, K, kk << 5, n0, lane);
      acc = __builtin_amdgcn_wmma_f32_16x16x32_f16(false, afr[kk], false, b, (short)0, acc,
                                                   false, false);
    }
    float bv = bias[n0 + n];
    float* orow = Out + (size_t)(m0 + mr) * NC + n0 + n;
#pragma unroll
    for (int r = 0; r < 8; ++r) {
      float v = acc[r] + bv;
      if (RELU) v = fmaxf(v, 0.0f);
      orow[(size_t)r * NC] = v;
    }
  }
}

// ---------------- fused edge MLP + scatter-add (avoids 640MB e_emb round trip) ---------------
// Per wave: 16 edges. e_emb_tile[16,128] via 16 WMMAs (A-frags register-resident),
// then msg = e_emb + x_node[col] scattered with global f32 atomics into ssum[row].
__global__ void edge_mlp_scatter(const float* __restrict__ edge_attr,
                                 const _Float16* __restrict__ WtE,  // [HID x IN_C] transposed
                                 const float* __restrict__ b_edge,
                                 const float* __restrict__ x_node,
                                 const int* __restrict__ ei,
                                 float* __restrict__ ssum) {
  int wave = blockIdx.x * (blockDim.x >> 5) + (threadIdx.x >> 5);
  int lane = threadIdx.x & 31;
  long long e0 = (long long)wave << 4;
  if (e0 >= (long long)N_EDGES) return;

  // speculative prefetch of a future tile's edge_attr (global_prefetch_b8)
  if (e0 + 256 < (long long)N_EDGES)
    __builtin_prefetch(edge_attr + (e0 + 256) * IN_C + (lane << 3), 0, 0);

  const float* atile = edge_attr + e0 * IN_C;
  v16h a0 = load_a_frag_f32(atile, IN_C, 0, lane);
  v16h a1 = load_a_frag_f32(atile, IN_C, 32, lane);

  int n  = lane & 15;
  int mr = (lane >> 4) << 3;
#pragma unroll
  for (int nt = 0; nt < 8; ++nt) {
    int n0 = nt << 4;
    v8f acc = {};
    v16h b0 = load_b_fragT(WtE, IN_C, 0, n0, lane);
    acc = __builtin_amdgcn_wmma_f32_16x16x32_f16(false, a0, false, b0, (short)0, acc,
                                                 false, false);
    v16h b1 = load_b_fragT(WtE, IN_C, 32, n0, lane);
    acc = __builtin_amdgcn_wmma_f32_16x16x32_f16(false, a1, false, b1, (short)0, acc,
                                                 false, false);
    int cn = n0 + n;
    float bv = b_edge[cn];
#pragma unroll
    for (int r = 0; r < 8; ++r) {
      long long e = e0 + mr + r;
      int dst  = ei[e];            // row = edge_index[0] (aggregate)
      int scol = ei[N_EDGES + e];  // col = edge_index[1] (gather)
      float val = acc[r] + bv + x_node[(size_t)scol * HID + cn];
      atomicAdd(ssum + (size_t)dst * HID + cn, val);
    }
  }
}

// ---------------- small helper kernels ----------------
__global__ void zero_kernel(float* __restrict__ p, long long n) {
  long long i = (long long)blockIdx.x * blockDim.x + threadIdx.x;
  long long s = (long long)gridDim.x * blockDim.x;
  for (; i < n; i += s) p[i] = 0.0f;
}

// f32 W[K x NC] -> f16 transposed Wt[NC x K]
__global__ void cvt_f16_transpose(const float* __restrict__ src, _Float16* __restrict__ dst,
                                  int K, int NC) {
  int i = blockIdx.x * blockDim.x + threadIdx.x;
  if (i >= K * NC) return;
  int k = i / NC, n = i - k * NC;
  dst[(size_t)n * K + k] = (_Float16)src[i];
}

// counts: cnt over row (= ei[0], message-pass mean), deg over dst (= ei[1], APPNP gcn_norm)
__global__ void count_deg(const int* __restrict__ ei, float* __restrict__ cnt,
                          float* __restrict__ deg) {
  long long e = (long long)blockIdx.x * blockDim.x + threadIdx.x;
  if (e >= (long long)N_EDGES) return;
  atomicAdd(cnt + ei[e], 1.0f);
  atomicAdd(deg + ei[N_EDGES + e], 1.0f);
}

__global__ void dinv_kernel(const float* __restrict__ deg, float* __restrict__ dinv) {
  int i = blockIdx.x * blockDim.x + threadIdx.x;
  if (i >= N_NODES) return;
  dinv[i] = rsqrtf(deg[i] + 1.0f);  // +1 = self loop; always > 0
}

__global__ void norm_kernel(const int* __restrict__ ei, const float* __restrict__ dinv,
                            float* __restrict__ norm) {
  long long e = (long long)blockIdx.x * blockDim.x + threadIdx.x;
  if (e >= (long long)N_EDGES) return;
  norm[e] = dinv[ei[e]] * dinv[ei[N_EDGES + e]];
}

// x_enh = relu((ssum + x_node) / (cnt + 1)); self loop folded in (x_node term, cnt+1). In place.
__global__ void finalize_enh(const float* __restrict__ x_node, const float* __restrict__ cnt,
                             float* __restrict__ ssum) {
  long long idx = (long long)blockIdx.x * blockDim.x + threadIdx.x;
  if (idx >= (long long)N_NODES * HID) return;
  int i = (int)(idx >> 7);
  float v = (ssum[idx] + x_node[idx]) / (cnt[i] + 1.0f);
  ssum[idx] = fmaxf(v, 0.0f);
}

// agg[dst] += norm_e * h[src]; 64 consecutive channel-threads per edge (coalesced)
__global__ void appnp_scatter(const int* __restrict__ ei, const float* __restrict__ norm,
                              const float* __restrict__ h, float* __restrict__ agg) {
  long long t = (long long)blockIdx.x * blockDim.x + threadIdx.x;
  long long e = t >> 6;
  int c = (int)(t & 63);
  if (e >= (long long)N_EDGES) return;
  int src = ei[e];            // APPNP: gather from edge_index[0]
  int dst = ei[N_EDGES + e];  //        scatter to  edge_index[1]
  atomicAdd(agg + (size_t)dst * OUT_C + c, norm[e] * h[(size_t)src * OUT_C + c]);
}

// h_next = 0.9*(agg + dinv[i]^2 * h_prev) + 0.1*h0   (self-loop term folded in)
__global__ void appnp_update(const float* __restrict__ agg, const float* __restrict__ hprev,
                             const float* __restrict__ h0, const float* __restrict__ dinv,
                             float* __restrict__ hnext) {
  long long idx = (long long)blockIdx.x * blockDim.x + threadIdx.x;
  if (idx >= (long long)N_NODES * OUT_C) return;
  int i = (int)(idx >> 6);
  float di = dinv[i];
  hnext[idx] = (1.0f - ALPHA_F) * (agg[idx] + di * di * hprev[idx]) + ALPHA_F * h0[idx];
}

// ---------------- host-side orchestration ----------------
extern "C" void kernel_launch(void* const* d_in, const int* in_sizes, int n_in,
                              void* d_out, int out_size, void* d_ws, size_t ws_size,
                              hipStream_t stream) {
  const float* x      = (const float*)d_in[0];
  const float* eattr  = (const float*)d_in[1];
  const float* W_node = (const float*)d_in[2];
  const float* b_node = (const float*)d_in[3];
  const float* W_edge = (const float*)d_in[4];
  const float* b_edge = (const float*)d_in[5];
  const float* W1     = (const float*)d_in[6];
  const float* b1     = (const float*)d_in[7];
  const float* W2     = (const float*)d_in[8];
  const float* b2     = (const float*)d_in[9];
  const int*   ei     = (const int*)d_in[10];
  float* out = (float*)d_out;

  // workspace layout (floats)
  float* ws  = (float*)d_ws;
  float* XN  = ws;                                 // [N,128] x_node, reused as h
  float* SS  = XN + (size_t)N_NODES * HID;         // [N,128] ssum -> x_enh (in place)
  float* CNT = SS + (size_t)N_NODES * HID;         // [N]
  float* DEG = CNT + N_NODES;                      // [N]
  float* DNV = DEG + N_NODES;                      // [N]
  float* H0  = DNV + N_NODES;                      // [N,64]
  float* B1  = H0 + (size_t)N_NODES * OUT_C;       // [N,64] ping-pong (d_out is the other)
  float* AGG = B1 + (size_t)N_NODES * OUT_C;       // [N,64]
  float* NRM = AGG + (size_t)N_NODES * OUT_C;      // [E]
  _Float16* WHn = (_Float16*)(NRM + N_EDGES);      // transposed [128 x 64]
  _Float16* WHe = WHn + IN_C * HID;                // transposed [128 x 64]
  _Float16* WH1 = WHe + IN_C * HID;                // transposed [128 x 128]
  _Float16* WH2 = WH1 + HID * HID;                 // transposed [64 x 128]

  // 1) weights -> f16, transposed (so B fragments load as contiguous b128)
  cvt_f16_transpose<<<(IN_C * HID + 255) / 256, 256, 0, stream>>>(W_node, WHn, IN_C, HID);
  cvt_f16_transpose<<<(IN_C * HID + 255) / 256, 256, 0, stream>>>(W_edge, WHe, IN_C, HID);
  cvt_f16_transpose<<<(HID * HID + 255) / 256, 256, 0, stream>>>(W1, WH1, HID, HID);
  cvt_f16_transpose<<<(HID * OUT_C + 255) / 256, 256, 0, stream>>>(W2, WH2, HID, OUT_C);

  // 2) zero accumulators (SS, CNT, DEG are contiguous)
  zero_kernel<<<2048, 256, 0, stream>>>(SS, (long long)N_NODES * HID + 2LL * N_NODES);

  // 3) x_node = x @ W_node + b_node      [100000 x 64 x 128]
  {
    int tiles_m = N_NODES / 16;
    gemm_wmma_f16<IN_C, HID, 0><<<(tiles_m + 7) / 8, 256, 0, stream>>>(x, WHn, b_node, XN,
                                                                       N_NODES);
  }

  // 4) degree counts (independent of GEMMs)
  count_deg<<<(N_EDGES + 255) / 256, 256, 0, stream>>>(ei, CNT, DEG);

  // 5) fused edge MLP + scatter-add into SS  [1.25M x 64 x 128 + 160M atomics]
  {
    int waves = N_EDGES / 16;  // 78125
    edge_mlp_scatter<<<(waves + 7) / 8, 256, 0, stream>>>(eattr, WHe, b_edge, XN, ei, SS);
  }

  // 6) gcn_norm pieces
  dinv_kernel<<<(N_NODES + 255) / 256, 256, 0, stream>>>(DEG, DNV);
  norm_kernel<<<(N_EDGES + 255) / 256, 256, 0, stream>>>(ei, DNV, NRM);

  // 7) x_enh = relu(mean) in place in SS
  finalize_enh<<<((long long)N_NODES * HID + 255) / 256, 256, 0, stream>>>(XN, CNT, SS);

  // 8) h = relu(x_enh @ W1 + b1) -> XN   [100000 x 128 x 128]
  {
    int tiles_m = N_NODES / 16;
    gemm_wmma_f16<HID, HID, 1><<<(tiles_m + 7) / 8, 256, 0, stream>>>(SS, WH1, b1, XN,
                                                                      N_NODES);
  }

  // 9) h0 = h @ W2 + b2 -> H0            [100000 x 128 x 64]
  {
    int tiles_m = N_NODES / 16;
    gemm_wmma_f16<HID, OUT_C, 0><<<(tiles_m + 7) / 8, 256, 0, stream>>>(XN, WH2, b2, H0,
                                                                        N_NODES);
  }

  // 10) APPNP: 10 iterations, ping-pong B1 <-> d_out (iter 9 writes d_out)
  const float* prev = H0;
  float* next = B1;
  long long scat_threads = (long long)N_EDGES * OUT_C;
  long long upd_threads  = (long long)N_NODES * OUT_C;
  for (int it = 0; it < K_ITER; ++it) {
    zero_kernel<<<2048, 256, 0, stream>>>(AGG, (long long)N_NODES * OUT_C);
    appnp_scatter<<<(unsigned)((scat_threads + 255) / 256), 256, 0, stream>>>(ei, NRM, prev, AGG);
    appnp_update<<<(unsigned)((upd_threads + 255) / 256), 256, 0, stream>>>(AGG, prev, H0, DNV, next);
    prev = next;
    next = (next == B1) ? out : B1;
  }
}
